// AdaptiveSoftmax_80461917323672
// MI455X (gfx1250) — matrix operand order repaired and checked by reading, not compiled, not run on hardware
//
#include <hip/hip_runtime.h>
#include <hip/hip_bf16.h>

typedef __attribute__((ext_vector_type(16))) __bf16 v16bf;
typedef __attribute__((ext_vector_type(8)))  float  v8f;

#define BROWS 4096
#define KIN   1024
#define CUT0  2000
#define CUT1  10000
#define MAXSPLIT 16

union FragU { v16bf v; uint4 q[2]; };
union AccU  { v8f v; float f[8]; };

// ---------------- data prep ----------------

__global__ void cvt_bf16_kernel(const float* __restrict__ src,
                                __hip_bfloat16* __restrict__ dst, int n) {
    int i = blockIdx.x * blockDim.x + threadIdx.x;
    if (i < n) dst[i] = __float2bfloat16(src[i]);
}

// W is [K x N] f32 row-major; WT is [Npad x K] bf16 (rows = columns of W, zero pad)
__global__ void transpose_bf16_kernel(const float* __restrict__ W,
                                      __hip_bfloat16* __restrict__ WT,
                                      int K, int N, int Npad) {
    int n = blockIdx.x * blockDim.x + threadIdx.x;
    int k = blockIdx.y;
    if (n >= Npad || k >= K) return;
    float v = (n < N) ? W[(size_t)k * N + n] : 0.0f;
    WT[(size_t)n * K + k] = __float2bfloat16(v);
}

// ---------------- 64-row GEMM, bf16 out: H = X @ W1 ----------------
// Wave tile 64x32: 8 accumulators, 4 A frags x 2 B frags per k-step.
__global__ __launch_bounds__(256) void gemm64_bf16out_kernel(
    const __hip_bfloat16* __restrict__ X,   // [4096 x K] bf16
    const __hip_bfloat16* __restrict__ WT,  // [N x K] bf16
    __hip_bfloat16* __restrict__ Hout,      // [4096 x N] bf16
    int K, int N) {
    extern __shared__ __align__(16) char smem_raw[];
    __hip_bfloat16* As = (__hip_bfloat16*)smem_raw;   // 64 x K
    const int tid  = threadIdx.x;
    const int lane = tid & 31;
    const int wave = tid >> 5;
    const int mblk = blockIdx.x;            // 64 rows
    const __hip_bfloat16* Arow = X + (size_t)mblk * 64 * K;
    for (int idx = tid * 8; idx < 64 * K; idx += 256 * 8)
        *(uint4*)(&As[idx]) = *(const uint4*)(&Arow[idx]);
    __syncthreads();

    const int mlo = lane & 15;
    const int hi  = lane >> 4;
    const int kb  = hi * 8;
    const int ntiles = N >> 5;              // 32-col tiles
    for (int nt = wave; nt < ntiles; nt += 8) {
        AccU accA0, accA1, accA2, accA3, accB0, accB1, accB2, accB3;
        accA0.v = (v8f){}; accA1.v = (v8f){}; accA2.v = (v8f){}; accA3.v = (v8f){};
        accB0.v = (v8f){}; accB1.v = (v8f){}; accB2.v = (v8f){}; accB3.v = (v8f){};
        const __hip_bfloat16* Bp = WT + (size_t)(nt * 32 + mlo) * K + hi * 8;
        const __hip_bfloat16* Bq = Bp + (size_t)16 * K;
        const __hip_bfloat16* A0 = &As[(size_t)mlo * K + kb];
        for (int k0 = 0; k0 < K; k0 += 32) {
            FragU b0, b1, a0, a1, a2, a3;
            b0.q[0] = *(const uint4*)(&Bp[k0]);
            b0.q[1] = *(const uint4*)(&Bp[k0 + 16]);
            b1.q[0] = *(const uint4*)(&Bq[k0]);
            b1.q[1] = *(const uint4*)(&Bq[k0 + 16]);
            a0.q[0] = *(const uint4*)(&A0[k0]);
            a0.q[1] = *(const uint4*)(&A0[k0 + 16]);
            a1.q[0] = *(const uint4*)(&A0[16 * K + k0]);
            a1.q[1] = *(const uint4*)(&A0[16 * K + k0 + 16]);
            a2.q[0] = *(const uint4*)(&A0[32 * K + k0]);
            a2.q[1] = *(const uint4*)(&A0[32 * K + k0 + 16]);
            a3.q[0] = *(const uint4*)(&A0[48 * K + k0]);
            a3.q[1] = *(const uint4*)(&A0[48 * K + k0 + 16]);
            accA0.v = __builtin_amdgcn_wmma_f32_16x16x32_bf16(false, a0.v, false, b0.v, (short)0, accA0.v, false, false);
            accA1.v = __builtin_amdgcn_wmma_f32_16x16x32_bf16(false, a1.v, false, b0.v, (short)0, accA1.v, false, false);
            accA2.v = __builtin_amdgcn_wmma_f32_16x16x32_bf16(false, a2.v, false, b0.v, (short)0, accA2.v, false, false);
            accA3.v = __builtin_amdgcn_wmma_f32_16x16x32_bf16(false, a3.v, false, b0.v, (short)0, accA3.v, false, false);
            accB0.v = __builtin_amdgcn_wmma_f32_16x16x32_bf16(false, a0.v, false, b1.v, (short)0, accB0.v, false, false);
            accB1.v = __builtin_amdgcn_wmma_f32_16x16x32_bf16(false, a1.v, false, b1.v, (short)0, accB1.v, false, false);
            accB2.v = __builtin_amdgcn_wmma_f32_16x16x32_bf16(false, a2.v, false, b1.v, (short)0, accB2.v, false, false);
            accB3.v = __builtin_amdgcn_wmma_f32_16x16x32_bf16(false, a3.v, false, b1.v, (short)0, accB3.v, false, false);
        }
        AccU* accs[8] = { &accA0, &accA1, &accA2, &accA3, &accB0, &accB1, &accB2, &accB3 };
        for (int half = 0; half < 2; ++half)
            for (int j = 0; j < 4; ++j)
                for (int r = 0; r < 8; ++r)
                    Hout[(size_t)(mblk * 64 + j * 16 + hi * 8 + r) * N +
                         nt * 32 + half * 16 + mlo] =
                        __float2bfloat16(accs[half * 4 + j]->f[r]);
    }
}

// ---- 64-row fused GEMM + online softmax partials + target gather, N-split ----
__global__ __launch_bounds__(256) void gemm64_lsm_kernel(
    const __hip_bfloat16* __restrict__ X,    // [4096 x K] bf16
    const __hip_bfloat16* __restrict__ WT,   // [Npad x K] bf16
    const int* __restrict__ target,
    float* __restrict__ pm, float* __restrict__ ps, float* __restrict__ pt,
    int K, int N, int Npad, int mode) {      // mode: 0 head, 1 tail0, 2 tail1
    extern __shared__ __align__(16) char smem_raw[];
    __hip_bfloat16* As = (__hip_bfloat16*)smem_raw;   // 64 x K
    __shared__ float red_m[8][64];
    __shared__ float red_s[8][64];
    __shared__ float tlog[64];
    __shared__ int   tcol[64];

    const int tid   = threadIdx.x;
    const int lane  = tid & 31;
    const int wave  = tid >> 5;
    const int mblk  = blockIdx.x;
    const int split = blockIdx.y;
    const int nsplit = gridDim.y;

    if (tid < 64) {
        int t = target[mblk * 64 + tid];
        int gi;
        if (mode == 0) {
            int cid = (t >= CUT0) + (t >= CUT1);
            gi = (cid == 0) ? t : (CUT0 + cid - 1);
        } else {
            int cs = (mode == 1) ? CUT0 : CUT1;
            gi = t - cs;
            gi = gi < 0 ? 0 : gi;
            gi = gi > N - 1 ? N - 1 : gi;
        }
        tcol[tid] = gi;
        tlog[tid] = 0.0f;
    }
    const __hip_bfloat16* Arow = X + (size_t)mblk * 64 * K;
    for (int idx = tid * 8; idx < 64 * K; idx += 256 * 8)
        *(uint4*)(&As[idx]) = *(const uint4*)(&Arow[idx]);
    __syncthreads();

    const int mlo = lane & 15;
    const int hi  = lane >> 4;
    const int kb  = hi * 8;
    float rm[4][8], rs[4][8];
    for (int j = 0; j < 4; ++j)
        for (int r = 0; r < 8; ++r) { rm[j][r] = -__builtin_inff(); rs[j][r] = 0.0f; }

    const int ntiles = Npad >> 5;            // 32-col tiles
    for (int nt = split * 8 + wave; nt < ntiles; nt += nsplit * 8) {
        AccU accA0, accA1, accA2, accA3, accB0, accB1, accB2, accB3;
        accA0.v = (v8f){}; accA1.v = (v8f){}; accA2.v = (v8f){}; accA3.v = (v8f){};
        accB0.v = (v8f){}; accB1.v = (v8f){}; accB2.v = (v8f){}; accB3.v = (v8f){};
        const __hip_bfloat16* Bp = WT + (size_t)(nt * 32 + mlo) * K + hi * 8;
        const __hip_bfloat16* Bq = Bp + (size_t)16 * K;
        const __hip_bfloat16* A0 = &As[(size_t)mlo * K + kb];
        for (int k0 = 0; k0 < K; k0 += 32) {
            FragU b0, b1, a0, a1, a2, a3;
            b0.q[0] = *(const uint4*)(&Bp[k0]);
            b0.q[1] = *(const uint4*)(&Bp[k0 + 16]);
            b1.q[0] = *(const uint4*)(&Bq[k0]);
            b1.q[1] = *(const uint4*)(&Bq[k0 + 16]);
            a0.q[0] = *(const uint4*)(&A0[k0]);
            a0.q[1] = *(const uint4*)(&A0[k0 + 16]);
            a1.q[0] = *(const uint4*)(&A0[16 * K + k0]);
            a1.q[1] = *(const uint4*)(&A0[16 * K + k0 + 16]);
            a2.q[0] = *(const uint4*)(&A0[32 * K + k0]);
            a2.q[1] = *(const uint4*)(&A0[32 * K + k0 + 16]);
            a3.q[0] = *(const uint4*)(&A0[48 * K + k0]);
            a3.q[1] = *(const uint4*)(&A0[48 * K + k0 + 16]);
            accA0.v = __builtin_amdgcn_wmma_f32_16x16x32_bf16(false, a0.v, false, b0.v, (short)0, accA0.v, false, false);
            accA1.v = __builtin_amdgcn_wmma_f32_16x16x32_bf16(false, a1.v, false, b0.v, (short)0, accA1.v, false, false);
            accA2.v = __builtin_amdgcn_wmma_f32_16x16x32_bf16(false, a2.v, false, b0.v, (short)0, accA2.v, false, false);
            accA3.v = __builtin_amdgcn_wmma_f32_16x16x32_bf16(false, a3.v, false, b0.v, (short)0, accA3.v, false, false);
            accB0.v = __builtin_amdgcn_wmma_f32_16x16x32_bf16(false, a0.v, false, b1.v, (short)0, accB0.v, false, false);
            accB1.v = __builtin_amdgcn_wmma_f32_16x16x32_bf16(false, a1.v, false, b1.v, (short)0, accB1.v, false, false);
            accB2.v = __builtin_amdgcn_wmma_f32_16x16x32_bf16(false, a2.v, false, b1.v, (short)0, accB2.v, false, false);
            accB3.v = __builtin_amdgcn_wmma_f32_16x16x32_bf16(false, a3.v, false, b1.v, (short)0, accB3.v, false, false);
        }
        // epilogue: two 16-col subtiles
        AccU* accs[8] = { &accA0, &accA1, &accA2, &accA3, &accB0, &accB1, &accB2, &accB3 };
        for (int half = 0; half < 2; ++half) {
            const int n = nt * 32 + half * 16 + mlo;
            if (n < N) {
                for (int j = 0; j < 4; ++j)
                    for (int r = 0; r < 8; ++r) {
                        float x = accs[half * 4 + j]->f[r];
                        int m = j * 16 + hi * 8 + r;
                        float nm = fmaxf(rm[j][r], x);
                        rs[j][r] = rs[j][r] * __expf(rm[j][r] - nm) + __expf(x - nm);
                        rm[j][r] = nm;
                        if (n == tcol[m]) tlog[m] = x;   // unique writer per row
                    }
            }
        }
    }
    // combine across the 16 lanes of each half-wave (rows stay within half)
    for (int off = 1; off < 16; off <<= 1) {
        for (int j = 0; j < 4; ++j)
            for (int r = 0; r < 8; ++r) {
                float om = __shfl_xor(rm[j][r], off, 32);
                float os = __shfl_xor(rs[j][r], off, 32);
                float nm = fmaxf(rm[j][r], om);
                rs[j][r] = rs[j][r] * __expf(rm[j][r] - nm) + os * __expf(om - nm);
                rm[j][r] = nm;
            }
    }
    if (mlo == 0) {
        for (int j = 0; j < 4; ++j)
            for (int r = 0; r < 8; ++r) {
                red_m[wave][j * 16 + hi * 8 + r] = rm[j][r];
                red_s[wave][j * 16 + hi * 8 + r] = rs[j][r];
            }
    }
    __syncthreads();
    if (tid < 64) {
        float M = -__builtin_inff(), S = 0.0f;
        for (int w = 0; w < 8; ++w) {
            float om = red_m[w][tid], os = red_s[w][tid];
            float nm = fmaxf(M, om);
            S = S * __expf(M - nm) + os * __expf(om - nm);
            M = nm;
        }
        size_t row = (size_t)mblk * 64 + tid;
        pm[(size_t)split * BROWS + row] = M;
        ps[(size_t)split * BROWS + row] = S;
        pt[(size_t)split * BROWS + row] = tlog[tid];
    }
}

// ---------------- combine N-split partials ----------------
__global__ void combine_splits_kernel(const float* __restrict__ pm,
                                      const float* __restrict__ ps,
                                      const float* __restrict__ pt,
                                      int nsplit, float* __restrict__ rowout) {
    int row = blockIdx.x * blockDim.x + threadIdx.x;
    if (row >= BROWS) return;
    float M = -__builtin_inff(), S = 0.0f, T = 0.0f;
    for (int s = 0; s < nsplit; ++s) {
        float om = pm[(size_t)s * BROWS + row];
        float os = ps[(size_t)s * BROWS + row];
        float nm = fmaxf(M, om);
        S = S * __expf(M - nm) + os * __expf(om - nm);
        M = nm;
        T += pt[(size_t)s * BROWS + row];
    }
    rowout[row] = T - (M + __logf(S));
}

// ---------------- combine head + tails, loss ----------------
__global__ __launch_bounds__(1024) void finalize_kernel(
    const float* __restrict__ headv, const float* __restrict__ t0v,
    const float* __restrict__ t1v, const int* __restrict__ target,
    float* __restrict__ out) {
    __shared__ float ssum[1024];
    int tid = threadIdx.x;
    float acc = 0.0f;
    for (int i = tid; i < BROWS; i += 1024) {
        int t = target[i];
        int cid = (t >= CUT0) + (t >= CUT1);
        float v = headv[i];
        if (cid == 1) v += t0v[i];
        if (cid == 2) v += t1v[i];
        out[i] = v;
        acc += v;
    }
    ssum[tid] = acc;
    __syncthreads();
    for (int s = 512; s > 0; s >>= 1) {
        if (tid < s) ssum[tid] += ssum[tid + s];
        __syncthreads();
    }
    if (tid == 0) out[BROWS] = -ssum[0] / (float)BROWS;
}

// ---------------- launch ----------------

extern "C" void kernel_launch(void* const* d_in, const int* in_sizes, int n_in,
                              void* d_out, int out_size, void* d_ws, size_t ws_size,
                              hipStream_t stream) {
    const float* x    = (const float*)d_in[0];   // [4096 x 1024]
    const int*   tgt  = (const int*)d_in[1];     // [4096]
    const float* Wh   = (const float*)d_in[2];   // [1024 x 2002]
    const float* W1_0 = (const float*)d_in[3];   // [1024 x 256]
    const float* W2_0 = (const float*)d_in[4];   // [256 x 8000]
    const float* W1_1 = (const float*)d_in[5];   // [1024 x 64]
    const float* W2_1 = (const float*)d_in[6];   // [64 x 22000]
    float* out = (float*)d_out;                  // 4096 + 1

    const int NH = 2002, NHP = 2016, H0 = 256, H1 = 64;
    const int N0 = 8000, N1 = 22000, N1P = 22016;

    char* p = (char*)d_ws;
    auto take = [&](size_t bytes) {
        char* r = p;
        p += (bytes + 255) & ~(size_t)255;
        return r;
    };
    __hip_bfloat16* Xbf  = (__hip_bfloat16*)take((size_t)BROWS * KIN * 2);
    __hip_bfloat16* WhT  = (__hip_bfloat16*)take((size_t)NHP * KIN * 2);
    __hip_bfloat16* W10T = (__hip_bfloat16*)take((size_t)H0 * KIN * 2);
    __hip_bfloat16* W20T = (__hip_bfloat16*)take((size_t)N0 * H0 * 2);
    __hip_bfloat16* W11T = (__hip_bfloat16*)take((size_t)H1 * KIN * 2);
    __hip_bfloat16* W21T = (__hip_bfloat16*)take((size_t)N1P * H1 * 2);
    __hip_bfloat16* H0bf = (__hip_bfloat16*)take((size_t)BROWS * H0 * 2);
    __hip_bfloat16* H1bf = (__hip_bfloat16*)take((size_t)BROWS * H1 * 2);
    float* pm    = (float*)take((size_t)MAXSPLIT * BROWS * 4);
    float* psum  = (float*)take((size_t)MAXSPLIT * BROWS * 4);
    float* pt    = (float*)take((size_t)MAXSPLIT * BROWS * 4);
    float* headv = (float*)take((size_t)BROWS * 4);
    float* t0v   = (float*)take((size_t)BROWS * 4);
    float* t1v   = (float*)take((size_t)BROWS * 4);

    // convert X to bf16
    {
        int n = BROWS * KIN;
        cvt_bf16_kernel<<<n / 256, 256, 0, stream>>>(x, Xbf, n);
    }
    // transpose + convert weights (WT[n*K+k] = W[k*N+n], zero-padded rows)
    transpose_bf16_kernel<<<dim3((NHP + 255) / 256, KIN), 256, 0, stream>>>(Wh,   WhT,  KIN, NH, NHP);
    transpose_bf16_kernel<<<dim3((H0  + 255) / 256, KIN), 256, 0, stream>>>(W1_0, W10T, KIN, H0, H0);
    transpose_bf16_kernel<<<dim3((N0  + 255) / 256, H0 ), 256, 0, stream>>>(W2_0, W20T, H0,  N0, N0);
    transpose_bf16_kernel<<<dim3((H1  + 255) / 256, KIN), 256, 0, stream>>>(W1_1, W11T, KIN, H1, H1);
    transpose_bf16_kernel<<<dim3((N1P + 255) / 256, H1 ), 256, 0, stream>>>(W2_1, W21T, H1,  N1, N1P);

    // bottleneck projections (64 rows/block, dynamic LDS = 64*K*2 bytes)
    gemm64_bf16out_kernel<<<BROWS / 64, 256, 64 * KIN * 2, stream>>>(Xbf, W10T, H0bf, KIN, H0);
    gemm64_bf16out_kernel<<<BROWS / 64, 256, 64 * KIN * 2, stream>>>(Xbf, W11T, H1bf, KIN, H1);

    // head: fused GEMM + softmax partials, 4 N-splits
    gemm64_lsm_kernel<<<dim3(BROWS / 64, 4), 256, 64 * KIN * 2, stream>>>(
        Xbf, WhT, tgt, pm, psum, pt, KIN, NH, NHP, 0);
    combine_splits_kernel<<<BROWS / 256, 256, 0, stream>>>(pm, psum, pt, 4, headv);

    // tail0: 8 N-splits
    gemm64_lsm_kernel<<<dim3(BROWS / 64, 8), 256, 64 * H0 * 2, stream>>>(
        H0bf, W20T, tgt, pm, psum, pt, H0, N0, N0, 1);
    combine_splits_kernel<<<BROWS / 256, 256, 0, stream>>>(pm, psum, pt, 8, t0v);

    // tail1: 16 N-splits
    gemm64_lsm_kernel<<<dim3(BROWS / 64, 16), 256, 64 * H1 * 2, stream>>>(
        H1bf, W21T, tgt, pm, psum, pt, H1, N1, N1P, 2);
    combine_splits_kernel<<<BROWS / 256, 256, 0, stream>>>(pm, psum, pt, 16, t1v);

    finalize_kernel<<<1, 1024, 0, stream>>>(headv, t0v, t1v, tgt, out);
}
